// EGCL_51007031607343
// MI455X (gfx1250) — compile-verified, matrix operand order checked by reading, not run on hardware
//
#include <hip/hip_runtime.h>
#include <cmath>

typedef __bf16 bf16_t;
typedef __attribute__((ext_vector_type(16))) __bf16 v16bf;
typedef __attribute__((ext_vector_type(8)))  __bf16 v8bf;
typedef __attribute__((ext_vector_type(8)))  float  v8f;

static __device__ __forceinline__ bf16_t f2bf(float f) {
    unsigned u = __builtin_bit_cast(unsigned, f);
    unsigned r = (u + 0x7FFFu + ((u >> 16) & 1u)) >> 16;
    unsigned short s = (unsigned short)r;
    return __builtin_bit_cast(bf16_t, s);
}
static __device__ __forceinline__ float bf2f(bf16_t b) {
    unsigned short s = __builtin_bit_cast(unsigned short, b);
    unsigned u = ((unsigned)s) << 16;
    return __builtin_bit_cast(float, u);
}
static __device__ __forceinline__ float silu(float x) { return x / (1.0f + __expf(-x)); }
static __device__ __forceinline__ float sigmoidf_(float x) { return 1.0f / (1.0f + __expf(-x)); }

// ---------------------------------------------------------------------------
// Load a 16x32 bf16 A-fragment from LDS per the ISA 16-bit layout:
// lanes<16 hold K{0..7,16..23}, lanes>=16 hold K{8..15,24..31} of row (lane&15).
// ---------------------------------------------------------------------------
static __device__ __forceinline__ v16bf load_frag(const bf16_t* __restrict__ row)
{
    v8bf lo = *(const v8bf*)(row);
    v8bf hi = *(const v8bf*)(row + 16);
    return __builtin_shufflevector(lo, hi, 0,1,2,3,4,5,6,7,8,9,10,11,12,13,14,15);
}

// ---------------------------------------------------------------------------
// Fused 128xK @ Kx128 WMMA GEMM helper. 8 waves; wave w owns rows [16w,16w+16).
// A in LDS (bf16, stride in halves, K multiple of 32). Bt in global: [128][strideB]
// bf16, row n = output column n (transposed weights). ACT: 0=none, 1=silu.
// MODE: 0 -> bf16 to LDS (strideO); 1 -> f32 to global + residual, guarded rows.
// ---------------------------------------------------------------------------
template<int ACT, int MODE>
static __device__ __forceinline__ void wmma_gemm_128(
    const bf16_t* __restrict__ Alds, int strideA, int K,
    const bf16_t* __restrict__ BtG, int strideB,
    const float* __restrict__ bias,
    bf16_t* __restrict__ outLds, int strideO,
    float* __restrict__ gOut, const float* __restrict__ gRes, int ldO, int validRows)
{
    const int lane  = threadIdx.x & 31;
    const int wave  = threadIdx.x >> 5;
    const int nloc  = lane & 15;
    const int mrow  = (wave << 4) + nloc;
    const int koff  = (lane < 16) ? 0 : 8;
    v8f acc[8] = {};
    for (int kb = 0; kb < K; kb += 32) {
        v16bf a = load_frag(Alds + mrow * strideA + kb + koff);
        #pragma unroll
        for (int t = 0; t < 8; ++t) {
            v16bf b = load_frag(BtG + ((t << 4) + nloc) * strideB + kb + koff);
            acc[t] = __builtin_amdgcn_wmma_f32_16x16x32_bf16(
                         false, a, false, b, (short)0, acc[t], false, false);
        }
    }
    const int mbase = (wave << 4) + ((lane >= 16) ? 8 : 0);
    #pragma unroll
    for (int t = 0; t < 8; ++t) {
        const int ncol = (t << 4) + nloc;
        const float bv = bias ? bias[ncol] : 0.0f;
        #pragma unroll
        for (int r = 0; r < 8; ++r) {
            const int m = mbase + r;
            float x = acc[t][r] + bv;
            if (ACT == 1) x = silu(x);
            if (MODE == 0) {
                outLds[m * strideO + ncol] = f2bf(x);
            } else {
                if (m < validRows) {
                    float res = gRes ? gRes[m * ldO + ncol] : 0.0f;
                    gOut[m * ldO + ncol] = x + res;
                }
            }
        }
    }
}

// ---------------------------------------------------------------------------
// Head GEMM: per-wave 16 rows x single 16-col N-tile (K multiple of 32).
// Returns the f32 accumulator; lane holds column (lane&15), rows mbase..mbase+7.
// ---------------------------------------------------------------------------
static __device__ __forceinline__ v8f wmma_head_16(
    const bf16_t* __restrict__ Alds, int strideA, int K,
    const bf16_t* __restrict__ BtG, int strideB)
{
    const int lane  = threadIdx.x & 31;
    const int wave  = threadIdx.x >> 5;
    const int nloc  = lane & 15;
    const int mrow  = (wave << 4) + nloc;
    const int koff  = (lane < 16) ? 0 : 8;
    v8f acc = {};
    for (int kb = 0; kb < K; kb += 32) {
        v16bf a = load_frag(Alds + mrow * strideA + kb + koff);
        v16bf b = load_frag(BtG + nloc * strideB + kb + koff);
        acc = __builtin_amdgcn_wmma_f32_16x16x32_bf16(
                  false, a, false, b, (short)0, acc, false, false);
    }
    return acc;
}

// ---------------------------------------------------------------------------
// Prep: W [K][Nsrc] f32 -> Wt [Ndst][Kpad] bf16 (zero padded in both dims).
// ---------------------------------------------------------------------------
__global__ __launch_bounds__(256) void prep_wt_kernel(
    const float* __restrict__ W, bf16_t* __restrict__ Wt,
    int K, int Nsrc, int Ndst, int Kpad)
{
    int idx = blockIdx.x * 256 + threadIdx.x;
    if (idx >= Ndst * Kpad) return;
    int n = idx / Kpad, k = idx - n * Kpad;
    float v = (k < K && n < Nsrc) ? W[k * Nsrc + n] : 0.0f;
    Wt[idx] = f2bf(v);
}

__global__ __launch_bounds__(256) void zero_f32_kernel(float* __restrict__ p, int n)
{
    for (int i = blockIdx.x * 256 + threadIdx.x; i < n; i += gridDim.x * 256) p[i] = 0.0f;
}

// ---------------------------------------------------------------------------
// Edge kernel: 128 edges per block, fused phi_e + phi_x + phi_y + gate + scatter.
// ---------------------------------------------------------------------------
#define TE 128
__global__ __launch_bounds__(256) void egcl_edge_kernel(
    const float* __restrict__ pos, const float* __restrict__ vec,
    const float* __restrict__ nf,
    const int* __restrict__ snd, const int* __restrict__ rcv,
    const bf16_t* __restrict__ We0t, const float* __restrict__ be0,
    const bf16_t* __restrict__ We1t, const float* __restrict__ be1,
    const bf16_t* __restrict__ Wx0t, const float* __restrict__ bx0,
    const bf16_t* __restrict__ Wx1t, const float* __restrict__ bx1,
    const bf16_t* __restrict__ WxfT,
    const bf16_t* __restrict__ Wy0t, const float* __restrict__ by0,
    const bf16_t* __restrict__ Wy1t, const float* __restrict__ by1,
    const bf16_t* __restrict__ WyfT,
    const bf16_t* __restrict__ WinfT, const float* __restrict__ binf,
    float* __restrict__ msum, float* __restrict__ sx, float* __restrict__ sy,
    int E)
{
    // LDS carve (159232 B of the 320KB WGP pool)
    __shared__ __align__(16) unsigned char smem[159232];
    bf16_t* ef   = (bf16_t*)(smem);                  // [128][296]; later reused as m [128][136]
    bf16_t* bufA = (bf16_t*)(smem + 75776);          // [128][136]
    bf16_t* bufB = (bf16_t*)(smem + 110592);         // [128][136]
    float*  dxs  = (float*) (smem + 145408);         // [128][12]
    float*  dys  = (float*) (smem + 151552);         // [128][12]
    int*    sid  = (int*)   (smem + 157696);         // [128]
    int*    rid  = (int*)   (smem + 158208);         // [128]
    float*  gate = (float*) (smem + 158720);         // [128]

    const int tid  = threadIdx.x;
    const int e0   = blockIdx.x * TE;
    const int lane = tid & 31;
    const int wave = tid >> 5;
    const int nloc = lane & 15;
    const int mbase = (wave << 4) + ((lane >= 16) ? 8 : 0);

    if (tid < TE) {
        int ge = e0 + tid;
        sid[tid] = (ge < E) ? snd[ge] : 0;
        rid[tid] = (ge < E) ? rcv[ge] : 0;
    }
    __syncthreads();

    // geometry: dx/dy (normalized) + lx^2/ly^2 into ef, zero K-pad 264..287
    for (int i = tid; i < TE * 4; i += 256) {
        int e = i >> 2, v = i & 3;
        int s = sid[e], r = rid[e];
        float dpx[3], dpy[3], sumx = 1e-8f, sumy = 1e-8f;
        #pragma unroll
        for (int d = 0; d < 3; ++d) {
            float a = pos[(r * 4 + v) * 3 + d] - pos[(s * 4 + v) * 3 + d];
            float b = vec[(r * 4 + v) * 3 + d] - vec[(s * 4 + v) * 3 + d];
            dpx[d] = a; sumx += a * a;
            dpy[d] = b; sumy += b * b;
        }
        float lx = sqrtf(sumx), ly = sqrtf(sumy);
        float ix = 1.0f / (1.0f + lx), iy = 1.0f / (1.0f + ly);
        #pragma unroll
        for (int d = 0; d < 3; ++d) {
            dxs[e * 12 + v * 3 + d] = dpx[d] * ix;
            dys[e * 12 + v * 3 + d] = dpy[d] * iy;
        }
        ef[e * 296 + 256 + v] = f2bf(lx * lx);
        ef[e * 296 + 260 + v] = f2bf(ly * ly);
        if (v == 0)
            for (int p = 264; p < 288; ++p) ef[e * 296 + p] = f2bf(0.0f);
    }
    // feature gather: fully coalesced (per iteration all threads share one edge row)
    for (int i = tid; i < TE * 256; i += 256) {
        int e = i >> 8, c = i & 255;
        int node = (c < 128) ? sid[e] : rid[e];
        ef[e * 296 + c] = f2bf(nf[node * 128 + (c & 127)]);
    }
    __syncthreads();

    // phi_e
    wmma_gemm_128<1,0>(ef,   296, 288, We0t, 288, be0, bufA, 136, nullptr, nullptr, 0, 0);
    __syncthreads();
    bf16_t* m = ef;   // reuse ef region, stride 136
    wmma_gemm_128<1,0>(bufA, 136, 128, We1t, 128, be1, m,    136, nullptr, nullptr, 0, 0);
    __syncthreads();

    // phi_x torso + WMMA head + scatter into shifts_x
    wmma_gemm_128<1,0>(m,    136, 128, Wx0t, 128, bx0, bufB, 136, nullptr, nullptr, 0, 0);
    __syncthreads();
    wmma_gemm_128<1,0>(bufB, 136, 128, Wx1t, 128, bx1, bufA, 136, nullptr, nullptr, 0, 0);
    __syncthreads();
    {
        v8f acc = wmma_head_16(bufA, 136, 128, WxfT, 128);
        if (nloc < 4) {
            const int v = nloc;
            #pragma unroll
            for (int r = 0; r < 8; ++r) {
                int e = mbase + r;
                if (e0 + e < E) {
                    float p = tanhf(acc[r]);
                    int rr = rid[e];
                    #pragma unroll
                    for (int d = 0; d < 3; ++d)
                        atomicAdd(&sx[rr * 12 + v * 3 + d], dxs[e * 12 + v * 3 + d] * p);
                }
            }
        }
    }
    __syncthreads();

    // phi_y torso + WMMA head + scatter into shifts_y
    wmma_gemm_128<1,0>(m,    136, 128, Wy0t, 128, by0, bufB, 136, nullptr, nullptr, 0, 0);
    __syncthreads();
    wmma_gemm_128<1,0>(bufB, 136, 128, Wy1t, 128, by1, bufA, 136, nullptr, nullptr, 0, 0);
    __syncthreads();
    {
        v8f acc = wmma_head_16(bufA, 136, 128, WyfT, 128);
        if (nloc < 4) {
            const int v = nloc;
            #pragma unroll
            for (int r = 0; r < 8; ++r) {
                int e = mbase + r;
                if (e0 + e < E) {
                    float p = tanhf(acc[r]);
                    int rr = rid[e];
                    #pragma unroll
                    for (int d = 0; d < 3; ++d)
                        atomicAdd(&sy[rr * 12 + v * 3 + d], dys[e * 12 + v * 3 + d] * p);
                }
            }
        }
    }

    // attention gate via WMMA head (column 0) + scatter gated messages
    {
        v8f acc = wmma_head_16(m, 136, 128, WinfT, 128);
        if (nloc == 0) {
            const float b0 = binf[0];
            #pragma unroll
            for (int r = 0; r < 8; ++r) {
                int e = mbase + r;
                gate[e] = sigmoidf_(acc[r] + b0);
            }
        }
    }
    __syncthreads();
    for (int i = tid; i < TE * 128; i += 256) {
        int e = i >> 7, f = i & 127;
        if (e0 + e >= E) continue;
        atomicAdd(&msum[rid[e] * 128 + f], bf2f(m[e * 136 + f]) * gate[e]);
    }
}

// ---------------------------------------------------------------------------
// Node kernel: phi_h MLP + all residual outputs.
// ---------------------------------------------------------------------------
#define TN 128
__global__ __launch_bounds__(256) void egcl_node_kernel(
    const float* __restrict__ msum, const float* __restrict__ nf,
    const float* __restrict__ pos, const float* __restrict__ vec,
    const float* __restrict__ sx, const float* __restrict__ sy,
    const bf16_t* __restrict__ Wh0t, const float* __restrict__ bh0,
    const bf16_t* __restrict__ Wh1t, const float* __restrict__ bh1,
    const bf16_t* __restrict__ Wh2t, const float* __restrict__ bh2,
    float* __restrict__ out, int N, float inv_avg)
{
    __shared__ __align__(16) unsigned char smem[137216];
    bf16_t* hin  = (bf16_t*)(smem);              // [128][264] (256 used)
    bf16_t* bufA = (bf16_t*)(smem + 67584);      // [128][136]
    bf16_t* bufB = (bf16_t*)(smem + 102400);     // [128][136]

    const int tid = threadIdx.x;
    const int n0  = blockIdx.x * TN;
    int valid = N - n0; if (valid > TN) valid = TN;

    float* out_pos  = out;
    float* out_vec  = out + (size_t)N * 12;
    float* out_feat = out + (size_t)N * 24;
    float* out_skip = out + (size_t)N * 24 + (size_t)N * 128;

    // stage hin = [m_i | node_features]
    for (int i = tid; i < TN * 256; i += 256) {
        int nl = i >> 8, c = i & 255;
        float v = 0.0f;
        if (nl < valid) {
            int n = n0 + nl;
            v = (c < 128) ? msum[n * 128 + c] * inv_avg : nf[n * 128 + (c - 128)];
        }
        hin[nl * 264 + c] = f2bf(v);
    }
    // vector-channel residual outputs
    for (int i = tid; i < TN * 12; i += 256) {
        int nl = i / 12, c = i - nl * 12;
        if (nl < valid) {
            int n = n0 + nl;
            float shx = sx[n * 12 + c] * inv_avg;
            float shy = sy[n * 12 + c] * inv_avg;
            out_pos[n * 12 + c]  = shx + pos[n * 12 + c];
            out_vec[n * 12 + c]  = shy + vec[n * 12 + c];
            out_skip[n * 12 + c] = shy;
        }
    }
    __syncthreads();

    wmma_gemm_128<1,0>(hin,  264, 256, Wh0t, 256, bh0, bufA, 136, nullptr, nullptr, 0, 0);
    __syncthreads();
    wmma_gemm_128<1,0>(bufA, 136, 128, Wh1t, 128, bh1, bufB, 136, nullptr, nullptr, 0, 0);
    __syncthreads();
    wmma_gemm_128<0,1>(bufB, 136, 128, Wh2t, 128, bh2, nullptr, 0,
                       out_feat + (size_t)n0 * 128, nf + (size_t)n0 * 128, 128, valid);
}

// ---------------------------------------------------------------------------
extern "C" void kernel_launch(void* const* d_in, const int* in_sizes, int n_in,
                              void* d_out, int out_size, void* d_ws, size_t ws_size,
                              hipStream_t stream)
{
    const float* pos = (const float*)d_in[0];
    const float* vec = (const float*)d_in[1];
    const float* nf  = (const float*)d_in[2];
    const int*   snd = (const int*)d_in[3];
    const int*   rcv = (const int*)d_in[4];
    const float* We0 = (const float*)d_in[5];  const float* be0 = (const float*)d_in[6];
    const float* We1 = (const float*)d_in[7];  const float* be1 = (const float*)d_in[8];
    const float* Winf= (const float*)d_in[9];  const float* binf= (const float*)d_in[10];
    const float* Wx0 = (const float*)d_in[11]; const float* bx0 = (const float*)d_in[12];
    const float* Wx1 = (const float*)d_in[13]; const float* bx1 = (const float*)d_in[14];
    const float* Wxf = (const float*)d_in[15];
    const float* Wy0 = (const float*)d_in[16]; const float* by0 = (const float*)d_in[17];
    const float* Wy1 = (const float*)d_in[18]; const float* by1 = (const float*)d_in[19];
    const float* Wyf = (const float*)d_in[20];
    const float* Wh0 = (const float*)d_in[21]; const float* bh0 = (const float*)d_in[22];
    const float* Wh1 = (const float*)d_in[23]; const float* bh1 = (const float*)d_in[24];
    const float* Wh2 = (const float*)d_in[25]; const float* bh2 = (const float*)d_in[26];
    (void)n_in; (void)out_size; (void)ws_size;

    const int N = in_sizes[2] / 128;
    const int E = in_sizes[3];
    const int avg = (int)std::sqrt((double)E);
    const float inv_avg = 1.0f / (float)avg;

    // workspace carve (256B aligned regions)
    unsigned char* ws = (unsigned char*)d_ws;
    size_t off = 0;
    auto carve = [&](size_t bytes) {
        unsigned char* p = ws + off;
        off = (off + bytes + 255) & ~(size_t)255;
        return p;
    };
    bf16_t* We0t = (bf16_t*)carve((size_t)128 * 288 * 2);
    bf16_t* We1t = (bf16_t*)carve((size_t)128 * 128 * 2);
    bf16_t* Wx0t = (bf16_t*)carve((size_t)128 * 128 * 2);
    bf16_t* Wx1t = (bf16_t*)carve((size_t)128 * 128 * 2);
    bf16_t* Wy0t = (bf16_t*)carve((size_t)128 * 128 * 2);
    bf16_t* Wy1t = (bf16_t*)carve((size_t)128 * 128 * 2);
    bf16_t* Wh0t = (bf16_t*)carve((size_t)128 * 256 * 2);
    bf16_t* Wh1t = (bf16_t*)carve((size_t)128 * 128 * 2);
    bf16_t* Wh2t = (bf16_t*)carve((size_t)128 * 128 * 2);
    bf16_t* WxfT = (bf16_t*)carve((size_t)16 * 128 * 2);
    bf16_t* WyfT = (bf16_t*)carve((size_t)16 * 128 * 2);
    bf16_t* WinfT= (bf16_t*)carve((size_t)16 * 128 * 2);
    float*  msum = (float*)carve((size_t)N * 128 * 4);
    float*  sx   = (float*)carve((size_t)N * 12 * 4);
    float*  sy   = (float*)carve((size_t)N * 12 * 4);

    auto launch_wt = [&](const float* W, bf16_t* Wt, int K, int Nsrc, int Ndst, int Kpad) {
        int total = Ndst * Kpad;
        prep_wt_kernel<<<(total + 255) / 256, 256, 0, stream>>>(W, Wt, K, Nsrc, Ndst, Kpad);
    };
    launch_wt(We0, We0t, 264, 128, 128, 288);
    launch_wt(We1, We1t, 128, 128, 128, 128);
    launch_wt(Wx0, Wx0t, 128, 128, 128, 128);
    launch_wt(Wx1, Wx1t, 128, 128, 128, 128);
    launch_wt(Wy0, Wy0t, 128, 128, 128, 128);
    launch_wt(Wy1, Wy1t, 128, 128, 128, 128);
    launch_wt(Wh0, Wh0t, 256, 128, 128, 256);
    launch_wt(Wh1, Wh1t, 128, 128, 128, 128);
    launch_wt(Wh2, Wh2t, 128, 128, 128, 128);
    launch_wt(Wxf, WxfT, 128, 4,  16, 128);
    launch_wt(Wyf, WyfT, 128, 4,  16, 128);
    launch_wt(Winf,WinfT,128, 1,  16, 128);

    zero_f32_kernel<<<512, 256, 0, stream>>>(msum, N * 128);
    zero_f32_kernel<<<128, 256, 0, stream>>>(sx, N * 12);
    zero_f32_kernel<<<128, 256, 0, stream>>>(sy, N * 12);

    egcl_edge_kernel<<<(E + TE - 1) / TE, 256, 0, stream>>>(
        pos, vec, nf, snd, rcv,
        We0t, be0, We1t, be1,
        Wx0t, bx0, Wx1t, bx1, WxfT,
        Wy0t, by0, Wy1t, by1, WyfT,
        WinfT, binf, msum, sx, sy, E);

    egcl_node_kernel<<<(N + TN - 1) / TN, 256, 0, stream>>>(
        msum, nf, pos, vec, sx, sy,
        Wh0t, bh0, Wh1t, bh1, Wh2t, bh2,
        (float*)d_out, N, inv_avg);
}
